// HybridConvQLSTM_65481071407483
// MI455X (gfx1250) — matrix operand order repaired and checked by reading, not compiled, not run on hardware
//
#include <hip/hip_runtime.h>
#include <hip/hip_bf16.h>

typedef __attribute__((ext_vector_type(16))) _Float16 v16h;
typedef __attribute__((ext_vector_type(8)))  float    v8f;
typedef __attribute__((ext_vector_type(4)))  unsigned u32x4;
typedef __attribute__((ext_vector_type(8)))  unsigned u32x8;

#define B_   64
#define T_   128
#define NH_  64
#define NC_  10
#define OH_  62
#define OW_  62

__device__ __forceinline__ float sigmoid_(float x) {
    return 1.0f / (1.0f + __expf(-x));
}

// ---------------------------------------------------------------------------
// WMMA wrapper: D = A(16x32 f16) * B(32x16 f16) + C(16x16 f32)
// ---------------------------------------------------------------------------
__device__ __forceinline__ v8f wmma_f16(v16h a, v16h b, v8f c) {
    return __builtin_amdgcn_wmma_f32_16x16x32_f16(
        /*neg_a=*/false, a, /*neg_b=*/false, b,
        /*c_mod=*/(short)0, c, /*reuse_a=*/false, /*reuse_b=*/false);
}

// ---------------------------------------------------------------------------
// Load a 16x32 f16 A-operand tile from an LDS row-major [16][64] f16 buffer.
// Per CDNA5 ISA 16-bit A layout.
// ---------------------------------------------------------------------------
__device__ __forceinline__ v16h loadA_lds(const _Float16* buf, int lane, int p) {
    const int hi = lane >> 4;
    const int ml = lane & 15;
    const _Float16* row = buf + ml * NH_ + p * 32;
    union { v16h h; unsigned u[8]; } a;
#pragma unroll
    for (int v = 0; v < 8; ++v) {
        const int k = (v < 4) ? (hi * 8 + 2 * v) : (16 + hi * 8 + 2 * (v - 4));
        a.u[v] = *(const unsigned*)(row + k);   // 2 contiguous f16 per dword
    }
    return a.h;
}

// ---------------------------------------------------------------------------
// Load a 32x16 f16 B-operand tile (Wt[k,n] = W[n,k]) from a global f32
// weight matrix W[64][stride]. n-block j, k-block p, koff shifts k origin.
// ---------------------------------------------------------------------------
__device__ __forceinline__ v16h loadB_w(const float* W, int lane, int j, int p,
                                        int stride, int koff) {
    const int hi = lane >> 4;
    const int lo = lane & 15;
    const int n  = 16 * j + lo;
    v16h b;
#pragma unroll
    for (int v = 0; v < 8; ++v) {
        const int k = 32 * p + ((v < 4) ? (hi * 8 + 2 * v) : (16 + hi * 8 + 2 * (v - 4)));
        b[2 * v]     = (_Float16)W[n * stride + koff + k];
        b[2 * v + 1] = (_Float16)W[n * stride + koff + k + 1];
    }
    return b;
}

// ===========================================================================
// Kernel 1: 3x3 VALID conv + sigmoid(x - thr) + spatial mean  -> feat[T][B]
// One workgroup per frame. The 16KB frame is DMA'd into LDS by the Tensor
// Data Mover (tensor_load_to_lds, TENSORcnt), freeing the waves during the
// fill; compute then runs entirely out of LDS so HBM is read exactly once.
// ===========================================================================
__global__ __launch_bounds__(256)
void conv_feat_kernel(const float* __restrict__ img,
                      const float* __restrict__ cw,
                      const float* __restrict__ cb,
                      const float* __restrict__ thr,
                      float* __restrict__ feat) {
    const int f   = blockIdx.x;          // frame = b*T + t
    const int tid = threadIdx.x;

    __shared__ float tileS[64 * 64];     // 16 KB frame
    __shared__ float red[256];

    // ---- TDM: DMA the whole frame into LDS (one wave owns TENSORcnt) ----
    if (tid < 32) {
        const unsigned long long gaddr =
            (unsigned long long)(const void*)(img + (size_t)f * 4096);
        // LDS aperture: low 32 bits of the generic pointer are the LDS offset
        const unsigned lds_addr = (unsigned)(unsigned long long)(void*)tileS;

        // D# group 0: count=1 | lds_addr | global_addr[56:0] | type=2
        u32x4 g0;
        g0[0] = 1u;                                           // count=1, is_restore=0
        g0[1] = lds_addr;                                     // bits [63:32]
        g0[2] = (unsigned)(gaddr & 0xffffffffu);              // global_addr[31:0]
        g0[3] = (unsigned)((gaddr >> 32) & 0x01ffffffu)       // global_addr[56:32]
              | (2u << 30);                                   // type=2 ("image")

        // D# group 1: 1 row of 4096 x 4-byte elements
        u32x8 g1;
        g1[0] = (2u << 16);          // workgroup_mask=0, data_size=2 (4 bytes)
        g1[1] = (4096u << 16);       // tensor_dim0[15:0]=4096 (atomic_barrier=0)
        g1[2] = (1u << 16);          // tensor_dim0[31:16]=0, tensor_dim1=1
        g1[3] = (4096u << 16);       // tensor_dim1[31:16]=0, tile_dim0=4096
        g1[4] = 1u;                  // tile_dim1=1, tile_dim2=0
        g1[5] = 4096u;               // tensor_dim0_stride[31:0]
        g1[6] = 0u;                  // stride[47:32]=0, tensor_dim1_stride lo
        g1[7] = 0u;

        u32x4 g2 = {0u, 0u, 0u, 0u};
        u32x4 g3 = {0u, 0u, 0u, 0u};

        asm volatile("tensor_load_to_lds %0, %1, %2, %3"
                     :
                     : "s"(g0), "s"(g1), "s"(g2), "s"(g3)
                     : "memory");
        __builtin_amdgcn_s_wait_tensorcnt(0);
    }

    const float k0 = cw[0], k1 = cw[1], k2 = cw[2];
    const float k3 = cw[3], k4 = cw[4], k5 = cw[5];
    const float k6 = cw[6], k7 = cw[7], k8 = cw[8];
    const float bias_m_thr = cb[0] - thr[0];

    __syncthreads();                     // LDS tile visible to all waves

    float sum = 0.0f;
    for (int o = tid; o < OH_ * OW_; o += 256) {
        const int oy = o / OW_;
        const int ox = o - oy * OW_;
        const float* p = tileS + oy * 64 + ox;
        float v = k0 * p[0]   + k1 * p[1]   + k2 * p[2]
                + k3 * p[64]  + k4 * p[65]  + k5 * p[66]
                + k6 * p[128] + k7 * p[129] + k8 * p[130]
                + bias_m_thr;
        sum += sigmoid_(v);
    }

    red[tid] = sum;
    __syncthreads();
#pragma unroll
    for (int s = 128; s > 0; s >>= 1) {
        if (tid < s) red[tid] += red[tid + s];
        __syncthreads();
    }
    if (tid == 0) {
        const int b = f >> 7;            // / T_
        const int t = f & 127;           // % T_
        feat[t * B_ + b] = red[0] * (1.0f / (float)(OH_ * OW_));
    }
}

// ===========================================================================
// Kernel 2: QLSTM recurrence + classifier.
// 4 workgroups x 16 batch rows. 8 waves = 4 gates x 2 column-halves.
// Weights held in VGPRs; activations exchanged via LDS f16 (WMMA A layout).
// ===========================================================================
struct QParams {
    const float* fcw[4]; const float* fcb[4];
    const float* lw[4];  const float* lb[4];
    const float* ew[4];  const float* eb[4];
    const float* clsw;   const float* clsb;
    const float* feat;   float* out;
};

__global__ __launch_bounds__(256)
void qlstm_kernel(QParams P) {
    const int tid  = threadIdx.x;
    const int lane = tid & 31;
    const int wv   = tid >> 5;
    const int g    = wv >> 1;            // 0=f 1=i 2=u 3=o
    const int jp   = wv & 1;             // column-half
    const int j0   = 2 * jp;
    const int j1   = j0 + 1;
    const int b0   = blockIdx.x * 16;

    __shared__ _Float16 zbuf[4][16 * NH_];   // fc outputs (f16, A layout)
    __shared__ _Float16 abuf[4][16 * NH_];   // tanh-layer outputs
    __shared__ float    gbuf[4][16 * NH_];   // final gate activations
    __shared__ _Float16 hxA[16 * NH_];       // hidden state f16 (A layout)
    __shared__ float    hxF[16 * NH_];       // hidden state f32 (classifier)
    __shared__ float    featL[T_ * 16];      // conv features for our rows
    __shared__ float    clsW[NC_ * NH_];
    __shared__ float    clsB[NC_];

    for (int e = tid; e < 16 * NH_; e += 256) { hxA[e] = (_Float16)0.0f; hxF[e] = 0.0f; }
    for (int e = tid; e < T_ * 16;  e += 256) {
        const int t = e >> 4, m = e & 15;
        featL[e] = P.feat[t * B_ + b0 + m];
    }
    for (int e = tid; e < NC_ * NH_; e += 256) clsW[e] = P.clsw[e];
    if (tid < NC_) clsB[tid] = P.clsb[tid];

    // -------- register-resident weight tiles (loop invariant) --------
    const float* fcw = P.fcw[g];
    const float* fcb = P.fcb[g];
    const float* lw  = P.lw[g];
    const float* lb  = P.lb[g];
    const float* ew  = P.ew[g];
    const float* eb  = P.eb[g];

    v16h fcB[2][2], lwB[2][2], ewB[2][2];    // [col-tile][k-block]
#pragma unroll
    for (int jj = 0; jj < 2; ++jj) {
        const int j = j0 + jj;
#pragma unroll
        for (int p = 0; p < 2; ++p) {
            fcB[jj][p] = loadB_w(fcw, lane, j, p, NH_ + 1, 1);  // hidden part
            lwB[jj][p] = loadB_w(lw,  lane, j, p, NH_, 0);
            ewB[jj][p] = loadB_w(ew,  lane, j, p, NH_, 0);
        }
    }

    // per-lane elementwise constants, keyed by n = 16*j + (lane&15)
    const int lo = lane & 15;
    const int hi = lane >> 4;
    float w0c[2], bf[2], bl[2], be[2];
#pragma unroll
    for (int jj = 0; jj < 2; ++jj) {
        const int n = 16 * (j0 + jj) + lo;
        w0c[jj] = fcw[n * (NH_ + 1)];    // x column of fc weight
        bf[jj]  = fcb[n];
        bl[jj]  = lb[n];
        be[jj]  = eb[n];
    }

    float cx[4] = {0.f, 0.f, 0.f, 0.f};
    __syncthreads();

    for (int t = 0; t < T_; ++t) {
        // ---- Stage 1: z = [xt|hx] @ fcW^T + fcB  (rank-1 x-term folded) ----
        v8f z0 = {}, z1 = {};
#pragma unroll
        for (int p = 0; p < 2; ++p) {
            const v16h A = loadA_lds(hxA, lane, p);
            z0 = wmma_f16(A, fcB[0][p], z0);
            z1 = wmma_f16(A, fcB[1][p], z1);
        }
#pragma unroll
        for (int r = 0; r < 8; ++r) {
            const int m  = r + 8 * hi;
            const float xv = featL[t * 16 + m];
            zbuf[g][m * NH_ + 16 * j0 + lo] = (_Float16)(z0[r] + xv * w0c[0] + bf[0]);
            zbuf[g][m * NH_ + 16 * j1 + lo] = (_Float16)(z1[r] + xv * w0c[1] + bf[1]);
        }
        __syncthreads();

        // ---- Stage 2: a = tanh(z @ lw^T + lb) ----
        v8f m10 = {}, m11 = {};
#pragma unroll
        for (int p = 0; p < 2; ++p) {
            const v16h A = loadA_lds(zbuf[g], lane, p);
            m10 = wmma_f16(A, lwB[0][p], m10);
            m11 = wmma_f16(A, lwB[1][p], m11);
        }
        float av0[8], av1[8];
#pragma unroll
        for (int r = 0; r < 8; ++r) {
            const int m = r + 8 * hi;
            av0[r] = tanhf(m10[r] + bl[0]);
            av1[r] = tanhf(m11[r] + bl[1]);
            abuf[g][m * NH_ + 16 * j0 + lo] = (_Float16)av0[r];
            abuf[g][m * NH_ + 16 * j1 + lo] = (_Float16)av1[r];
        }
        __syncthreads();

        // ---- Stage 3: gate = act(a * sigmoid(a @ ew^T + eb)) ----
        v8f m20 = {}, m21 = {};
#pragma unroll
        for (int p = 0; p < 2; ++p) {
            const v16h A = loadA_lds(abuf[g], lane, p);
            m20 = wmma_f16(A, ewB[0][p], m20);
            m21 = wmma_f16(A, ewB[1][p], m21);
        }
#pragma unroll
        for (int r = 0; r < 8; ++r) {
            const int m = r + 8 * hi;
            const float q0 = av0[r] * sigmoid_(m20[r] + be[0]);
            const float q1 = av1[r] * sigmoid_(m21[r] + be[1]);
            const float g0 = (g == 2) ? tanhf(q0) : sigmoid_(q0);
            const float g1 = (g == 2) ? tanhf(q1) : sigmoid_(q1);
            gbuf[g][m * NH_ + 16 * j0 + lo] = g0;
            gbuf[g][m * NH_ + 16 * j1 + lo] = g1;
        }
        __syncthreads();

        // ---- Stage 4: cell update (4 elements per thread) ----
#pragma unroll
        for (int c = 0; c < 4; ++c) {
            const int e = tid * 4 + c;
            const float fg = gbuf[0][e];
            const float ig = gbuf[1][e];
            const float ug = gbuf[2][e];
            const float og = gbuf[3][e];
            cx[c] = fg * cx[c] + ig * ug;
            const float h = og * tanhf(cx[c]);
            hxA[e] = (_Float16)h;
            hxF[e] = h;
        }
        __syncthreads();

        // ---- Stage 5: classifier out[t, b0+m, :] = hx @ cls^T + clsB ----
        if (tid < 16 * NC_) {
            const int m = tid / NC_;
            const int n = tid - m * NC_;
            float s = clsB[n];
#pragma unroll 8
            for (int k = 0; k < NH_; ++k)
                s += hxF[m * NH_ + k] * clsW[n * NH_ + k];
            P.out[(t * B_ + b0 + m) * NC_ + n] = s;
        }
        __syncthreads();
    }
}

// ===========================================================================
extern "C" void kernel_launch(void* const* d_in, const int* in_sizes, int n_in,
                              void* d_out, int out_size, void* d_ws, size_t ws_size,
                              hipStream_t stream) {
    const float* images = (const float*)d_in[0];
    const float* thr    = (const float*)d_in[1];
    const float* convw  = (const float*)d_in[2];
    const float* convb  = (const float*)d_in[3];

    float* feat = (float*)d_ws;                      // T_*B_ floats (32 KB)

    conv_feat_kernel<<<B_ * T_, 256, 0, stream>>>(images, convw, convb, thr, feat);

    QParams P;
    for (int gg = 0; gg < 4; ++gg) {
        const int base = 4 + 6 * gg;                 // setup_inputs dict order
        P.fcw[gg] = (const float*)d_in[base + 0];
        P.fcb[gg] = (const float*)d_in[base + 1];
        P.lw[gg]  = (const float*)d_in[base + 2];
        P.lb[gg]  = (const float*)d_in[base + 3];
        P.ew[gg]  = (const float*)d_in[base + 4];
        P.eb[gg]  = (const float*)d_in[base + 5];
    }
    P.clsw = (const float*)d_in[28];
    P.clsb = (const float*)d_in[29];
    P.feat = feat;
    P.out  = (float*)d_out;

    qlstm_kernel<<<B_ / 16, 256, 0, stream>>>(P);
}
